// GramMLPAttention_61186104099471
// MI455X (gfx1250) — matrix operand order repaired and checked by reading, not compiled, not run on hardware
//
#include <hip/hip_runtime.h>
#include <math.h>
#include <stdint.h>

// ---------------------------------------------------------------------------
// GramMLPAttention for MI455X (gfx1250), bf16 WMMA path, fp32 accumulate.
// QKV kept in bf16 end-to-end for the attention consumer; double-buffered
// LDS staging; b128 fragment loads; global_prefetch two tiles ahead.
// ---------------------------------------------------------------------------

typedef __bf16 bf16_t;
typedef __attribute__((ext_vector_type(16))) __bf16 v16bf;
typedef __attribute__((ext_vector_type(8)))  float  v8f;

#define DM    1024
#define NH    16
#define DH    64
#define TSEQ  2048
#define BB    2
#define NGRAM 21
#define DECAYF 0.99f
#define CHUNK 128
#define NCHUNK (TSEQ / CHUNK)

// A-fragment 16x32 bf16: row = lane&15; k runs contiguously for h<8 and h>=8.
__device__ __forceinline__ int kidxA(int h, int lane) {
  return ((h >> 3) << 4) + ((lane >> 4) << 3) + (h & 7);
}
// B-fragment 32x16 bf16: col = lane&15, k = kidxA(h, lane), same k order as A.

__device__ __forceinline__ uint32_t pack2(float a, float b) {
  union { bf16_t h[2]; uint32_t u; } cv;
  cv.h[0] = (bf16_t)a; cv.h[1] = (bf16_t)b;
  return cv.u;
}
__device__ __forceinline__ uint4 pack8(float4 a, float4 b) {
  uint4 r;
  r.x = pack2(a.x, a.y); r.y = pack2(a.z, a.w);
  r.z = pack2(b.x, b.y); r.w = pack2(b.z, b.w);
  return r;
}

// ===========================================================================
// GEMM: C[M,N] = A[M,K] @ W[K,N] (+bias). M,N multiples of 128, K of 32.
// 256 thr = 8 waves; block tile 128x128; wave tile 32x64 (8 WMMA / k-step).
// Double-buffered LDS; B tile transposed [n][k]. Output f32 or bf16.
// ===========================================================================
#define APITCH 40   // halves per row (32+8 pad) = 80B, 16B-aligned rows
#define BPITCH 40

__global__ __launch_bounds__(256)
void gemm_bf16_wmma(const float* __restrict__ A, const float* __restrict__ Bw,
                    const float* __restrict__ bias, void* __restrict__ Cv,
                    int M, int N, int K, int hasBias, int outBf16)
{
  __shared__ bf16_t As [2][128 * APITCH];   // [row][k]
  __shared__ bf16_t Bst[2][128 * BPITCH];   // [n][k]
  const int tid  = threadIdx.x;
  const int lane = tid & 31;
  const int w    = tid >> 5;
  const int wm   = w & 3;
  const int wn   = w >> 2;
  const int bm   = blockIdx.y * 128;
  const int bn   = blockIdx.x * 128;
  const int lr   = lane & 15;
  const int hi   = lane >> 4;
  const int KT   = K >> 5;

  int ar[2], ac8[2], br2[2], bc4[2];
  #pragma unroll
  for (int i = 0; i < 2; ++i) {
    int slot = i * 256 + tid;
    ar[i]  = slot >> 2;  ac8[i] = (slot & 3) * 8;    // A: 128 rows x 4 c8
    br2[i] = slot >> 5;  bc4[i] = (slot & 31) * 4;   // B: 16 row-pairs x 32 c4
  }
  float4 fa[2][2], fb[2][2];

  auto fetch = [&](int kt) {
    const int k0 = kt << 5;
    #pragma unroll
    for (int i = 0; i < 2; ++i) {
      const float* ap = A + (size_t)(bm + ar[i]) * K + k0 + ac8[i];
      fa[i][0] = *(const float4*)ap;
      fa[i][1] = *(const float4*)(ap + 4);
      const float* bp = Bw + (size_t)(k0 + 2 * br2[i]) * N + bn + bc4[i];
      fb[i][0] = *(const float4*)bp;
      fb[i][1] = *(const float4*)(bp + N);
    }
    if (kt + 1 < KT) {   // prefetch two tiles ahead of compute
      const int k1 = (kt + 1) << 5;
      __builtin_prefetch(A + (size_t)(bm + ar[0]) * K + k1 + ac8[0], 0, 1);
      __builtin_prefetch(Bw + (size_t)(k1 + 2 * br2[0]) * N + bn + bc4[0], 0, 1);
    }
  };
  auto stash = [&](int buf) {
    #pragma unroll
    for (int i = 0; i < 2; ++i) {
      *(uint4*)&As[buf][ar[i] * APITCH + ac8[i]] = pack8(fa[i][0], fa[i][1]);
      #pragma unroll
      for (int j = 0; j < 4; ++j) {
        float e0 = ((const float*)&fb[i][0])[j];
        float e1 = ((const float*)&fb[i][1])[j];
        *(uint32_t*)&Bst[buf][(bc4[i] + j) * BPITCH + 2 * br2[i]] = pack2(e0, e1);
      }
    }
  };

  v8f acc[2][4] = {};
  fetch(0); stash(0); __syncthreads();

  for (int kt = 0; kt < KT; ++kt) {
    if (kt + 1 < KT) fetch(kt + 1);
    const int buf = kt & 1;
    v16bf a0, a1;
    #pragma unroll
    for (int h = 0; h < 16; ++h) {
      a0[h] = As[buf][(wm * 32 +      lr) * APITCH + kidxA(h, lane)];
      a1[h] = As[buf][(wm * 32 + 16 + lr) * APITCH + kidxA(h, lane)];
    }
    #pragma unroll
    for (int nt = 0; nt < 4; ++nt) {
      v16bf bfr;
      #pragma unroll
      for (int h = 0; h < 16; ++h)
        bfr[h] = Bst[buf][(wn * 64 + nt * 16 + lr) * BPITCH + kidxA(h, lane)];
      acc[0][nt] = __builtin_amdgcn_wmma_f32_16x16x32_bf16(
                     false, a0, false, bfr, (short)0, acc[0][nt], false, false);
      acc[1][nt] = __builtin_amdgcn_wmma_f32_16x16x32_bf16(
                     false, a1, false, bfr, (short)0, acc[1][nt], false, false);
    }
    if (kt + 1 < KT) stash((kt + 1) & 1);
    __syncthreads();
  }

  float bcol[4];
  #pragma unroll
  for (int nt = 0; nt < 4; ++nt)
    bcol[nt] = hasBias ? bias[bn + wn * 64 + nt * 16 + lr] : 0.f;

  const size_t rbase = (size_t)(bm + wm * 32 + (hi << 3)) * N + bn + wn * 64 + lr;
  if (outBf16) {
    bf16_t* cb = (bf16_t*)Cv + rbase;
    #pragma unroll
    for (int sub = 0; sub < 2; ++sub)
      #pragma unroll
      for (int r = 0; r < 8; ++r) {
        bf16_t* cp = cb + (size_t)(sub * 16 + r) * N;
        cp[0]  = (bf16_t)(acc[sub][0][r] + bcol[0]);
        cp[16] = (bf16_t)(acc[sub][1][r] + bcol[1]);
        cp[32] = (bf16_t)(acc[sub][2][r] + bcol[2]);
        cp[48] = (bf16_t)(acc[sub][3][r] + bcol[3]);
      }
  } else {
    float* cb = (float*)Cv + rbase;
    #pragma unroll
    for (int sub = 0; sub < 2; ++sub)
      #pragma unroll
      for (int r = 0; r < 8; ++r) {
        float* cp = cb + (size_t)(sub * 16 + r) * N;
        cp[0]  = acc[sub][0][r] + bcol[0];
        cp[16] = acc[sub][1][r] + bcol[1];
        cp[32] = acc[sub][2][r] + bcol[2];
        cp[48] = acc[sub][3][r] + bcol[3];
      }
  }
}

// ===========================================================================
// Flash attention (causal), bf16 WMMA, bf16 QKV input. 128 thr = 4 waves,
// 64 q-rows/block. Double-buffered K/V; Kt [key][d]; Vtt transposed [d][key].
// ===========================================================================
#define KPITCH 72   // 144B rows, 16B aligned
#define VPITCH 40
#define PPITCH 40

__global__ __launch_bounds__(128)
void flash_attn_wmma(const bf16_t* __restrict__ qkv, float* __restrict__ seq_out)
{
  __shared__ bf16_t Kt [2][32 * KPITCH];
  __shared__ bf16_t Vtt[2][64 * VPITCH];
  __shared__ bf16_t Pt[4][16 * PPITCH];

  const int tid = threadIdx.x, lane = tid & 31, w = tid >> 5;
  const int lr = lane & 15, hi = lane >> 4;
  const int qbase = blockIdx.x * 64;
  const int h = blockIdx.y;
  const int b = blockIdx.z;
  const size_t rowstride = 3 * DM;
  const size_t base = (size_t)b * TSEQ;

  // Q fragments: raw 16B loads (memory order == fragment order, no cvt)
  v16bf qf[2];
  {
    const bf16_t* qrowp = qkv + (base + qbase + w * 16 + lr) * rowstride + h * DH;
    union { v16bf v; uint4 q[2]; } qu;
    #pragma unroll
    for (int ks = 0; ks < 2; ++ks) {
      int d0 = ks * 32 + (hi << 3);
      qu.q[0] = *(const uint4*)(qrowp + d0);        // halves d0..d0+7
      qu.q[1] = *(const uint4*)(qrowp + d0 + 16);   // halves d0+16..d0+23
      qf[ks] = qu.v;
    }
  }

  // staging slots: K 32x(64/8)=256 slots; V 16 key-pairs x 16 d4 = 256 slots
  int kkr[2], kd8[2], vk2[2], vd4[2];
  #pragma unroll
  for (int i = 0; i < 2; ++i) {
    int slot = i * 128 + tid;
    kkr[i] = slot >> 3;  kd8[i] = (slot & 7) * 8;
    vk2[i] = slot >> 4;  vd4[i] = (slot & 15) * 4;
  }
  uint4 fk[2];
  uint2 fva[2], fvb[2];

  auto fetchKV = [&](int kv) {
    const int kv0 = kv * 32;
    #pragma unroll
    for (int i = 0; i < 2; ++i) {
      const bf16_t* kp = qkv + (base + kv0 + kkr[i]) * rowstride + h * DH + kd8[i] + DM;
      fk[i] = *(const uint4*)kp;
      const bf16_t* vp = qkv + (base + kv0 + 2 * vk2[i]) * rowstride + h * DH + vd4[i] + 2 * DM;
      fva[i] = *(const uint2*)vp;
      fvb[i] = *(const uint2*)(vp + rowstride);
    }
    if (kv0 + 64 < TSEQ) {   // prefetch two kv blocks ahead
      __builtin_prefetch(qkv + (base + kv0 + 32 + kkr[0]) * rowstride + h * DH + DM, 0, 1);
      __builtin_prefetch(qkv + (base + kv0 + 32 + 2 * vk2[0]) * rowstride + h * DH + 2 * DM, 0, 1);
    }
  };
  auto stashKV = [&](int buf) {
    #pragma unroll
    for (int i = 0; i < 2; ++i) {
      *(uint4*)&Kt[buf][kkr[i] * KPITCH + kd8[i]] = fk[i];
      // transpose bf16 pairs: {key even, key odd} per d
      uint32_t u0 = (fva[i].x & 0xffffu)  | (fvb[i].x << 16);
      uint32_t u1 = (fva[i].x >> 16)      | (fvb[i].x & 0xffff0000u);
      uint32_t u2 = (fva[i].y & 0xffffu)  | (fvb[i].y << 16);
      uint32_t u3 = (fva[i].y >> 16)      | (fvb[i].y & 0xffff0000u);
      bf16_t* vb = &Vtt[buf][vd4[i] * VPITCH + 2 * vk2[i]];
      *(uint32_t*)(vb)              = u0;
      *(uint32_t*)(vb + VPITCH)     = u1;
      *(uint32_t*)(vb + 2 * VPITCH) = u2;
      *(uint32_t*)(vb + 3 * VPITCH) = u3;
    }
  };

  v8f o[4] = {};
  float mi[8], li_[8];
  #pragma unroll
  for (int r = 0; r < 8; ++r) { mi[r] = -1e30f; li_[r] = 0.f; }

  const int nkv = (qbase + 64) / 32;
  fetchKV(0); stashKV(0); __syncthreads();

  for (int kv = 0; kv < nkv; ++kv) {
    if (kv + 1 < nkv) fetchKV(kv + 1);
    const int buf = kv & 1;
    const int kv0 = kv * 32;

    // S = Q @ K^T
    v8f s[2] = {};
    #pragma unroll
    for (int ks = 0; ks < 2; ++ks)
      #pragma unroll
      for (int nt = 0; nt < 2; ++nt) {
        v16bf bk;
        #pragma unroll
        for (int hh = 0; hh < 16; ++hh)
          bk[hh] = Kt[buf][(nt * 16 + lr) * KPITCH + ks * 32 + kidxA(hh, lane)];
        s[nt] = __builtin_amdgcn_wmma_f32_16x16x32_bf16(
                  false, qf[ks], false, bk, (short)0, s[nt], false, false);
      }

    // scale + causal mask + online softmax
    float p0[8], p1[8];
    #pragma unroll
    for (int r = 0; r < 8; ++r) {
      int qrow = qbase + w * 16 + r + (hi << 3);
      int key0 = kv0 + lr;
      float v0 = (key0      > qrow) ? -1e30f : s[0][r] * 0.125f;
      float v1 = (key0 + 16 > qrow) ? -1e30f : s[1][r] * 0.125f;
      float rowmax = fmaxf(v0, v1);
      #pragma unroll
      for (int m = 1; m < 16; m <<= 1)
        rowmax = fmaxf(rowmax, __shfl_xor(rowmax, m, 32));
      float mnew  = fmaxf(mi[r], rowmax);
      float alpha = __expf(mi[r] - mnew);
      float e0 = __expf(v0 - mnew);
      float e1 = __expf(v1 - mnew);
      float rs = e0 + e1;
      #pragma unroll
      for (int m = 1; m < 16; m <<= 1)
        rs += __shfl_xor(rs, m, 32);
      li_[r] = li_[r] * alpha + rs;
      mi[r]  = mnew;
      o[0][r] *= alpha; o[1][r] *= alpha; o[2][r] *= alpha; o[3][r] *= alpha;
      p0[r] = e0; p1[r] = e1;
    }

    // C-layout P -> per-wave LDS -> A-layout fragment (same-wave DS in-order)
    #pragma unroll
    for (int r = 0; r < 8; ++r) {
      int m = r + (hi << 3);
      Pt[w][m * PPITCH +      lr] = (bf16_t)p0[r];
      Pt[w][m * PPITCH + 16 + lr] = (bf16_t)p1[r];
    }
    __builtin_amdgcn_wave_barrier();

    v16bf ap;
    #pragma unroll
    for (int hh = 0; hh < 16; ++hh)
      ap[hh] = Pt[w][lr * PPITCH + kidxA(hh, lane)];
    #pragma unroll
    for (int nt = 0; nt < 4; ++nt) {
      v16bf bv;
      #pragma unroll
      for (int hh = 0; hh < 16; ++hh)
        bv[hh] = Vtt[buf][(nt * 16 + lr) * VPITCH + kidxA(hh, lane)];
      o[nt] = __builtin_amdgcn_wmma_f32_16x16x32_bf16(
                false, ap, false, bv, (short)0, o[nt], false, false);
    }

    if (kv + 1 < nkv) stashKV((kv + 1) & 1);
    __syncthreads();
  }

  #pragma unroll
  for (int nt = 0; nt < 4; ++nt)
    #pragma unroll
    for (int r = 0; r < 8; ++r) {
      int q = qbase + w * 16 + r + (hi << 3);
      int d = nt * 16 + lr;
      seq_out[(((size_t)b * NH + h) * TSEQ + q) * DH + d] = o[nt][r] / li_[r];
    }
}

// ===========================================================================
// Small elementwise / helper kernels
// ===========================================================================
__global__ void concat_w12(const float* __restrict__ w1w, const float* __restrict__ w2w,
                           float* __restrict__ wcat)
{
  int idx = blockIdx.x * blockDim.x + threadIdx.x;
  if (idx >= DM * DH) return;
  int row = idx / DH, col = idx % DH;
  wcat[(size_t)row * 128 + col]      = w1w[idx];
  wcat[(size_t)row * 128 + 64 + col] = w2w[idx];
}

__global__ void gate_kernel(const float* __restrict__ x, const float* __restrict__ gw,
                            const float* __restrict__ gb, float* __restrict__ gate)
{
  int idx = blockIdx.x * blockDim.x + threadIdx.x;
  if (idx >= BB * TSEQ * NH) return;
  int bt = idx >> 4, h = idx & 15;
  const float* xr = x + (size_t)bt * DM;
  float acc = gb[h];
  for (int k = 0; k < DM; k += 4) {
    float4 xv = *(const float4*)(xr + k);
    acc += xv.x * gw[k * NH + h]       + xv.y * gw[(k + 1) * NH + h]
         + xv.z * gw[(k + 2) * NH + h] + xv.w * gw[(k + 3) * NH + h];
  }
  gate[idx] = 1.f / (1.f + __expf(-acc));
}

__global__ void plucker_kernel(const float* __restrict__ w12x, float* __restrict__ wl)
{
  int idx = blockIdx.x * blockDim.x + threadIdx.x;   // (b*NH+h)*T + t
  if (idx >= BB * NH * TSEQ) return;
  int t  = idx % TSEQ;
  int bh = idx / TSEQ;
  int h = bh % NH, b = bh / NH;
  float p1[4], p2[4];
  #pragma unroll
  for (int c = 0; c < 4; ++c) {
    p1[c] = (t == 0) ? 0.f
          : w12x[(size_t)(b * TSEQ + t - 1) * 128 + h * 4 + c];
    p2[c] = w12x[(size_t)(b * TSEQ + t) * 128 + 64 + h * 4 + c];
  }
  float e[6] = { p1[0]*p2[1]-p1[1]*p2[0], p1[0]*p2[2]-p1[2]*p2[0],
                 p1[0]*p2[3]-p1[3]*p2[0], p1[1]*p2[2]-p1[2]*p2[1],
                 p1[1]*p2[3]-p1[3]*p2[1], p1[2]*p2[3]-p1[3]*p2[2] };
  float ss = 0.f;
  #pragma unroll
  for (int i = 0; i < 6; ++i) ss += e[i] * e[i];
  float inv = 1.f / fmaxf(sqrtf(ss), 1e-12f);
  #pragma unroll
  for (int i = 0; i < 6; ++i) wl[(size_t)idx * 6 + i] = e[i] * inv;
}

__global__ void gram_scan_local(const float* __restrict__ wl,
                                float* __restrict__ grams,
                                float* __restrict__ carry)
{
  int u = blockIdx.x * blockDim.x + threadIdx.x;
  const int NSEQ = BB * NH * NGRAM;
  if (u >= NSEQ * NCHUNK) return;
  int s = u / NCHUNK, c = u % NCHUNK;
  int p = s % NGRAM, bh = s / NGRAM;
  const int TI[NGRAM] = {0,0,0,0,0,0,1,1,1,1,1,2,2,2,2,3,3,3,4,4,5};
  const int TJ[NGRAM] = {0,1,2,3,4,5,1,2,3,4,5,2,3,4,5,3,4,5,4,5,5};
  int i = TI[p], j = TJ[p];
  const float* wlp = wl + (size_t)bh * TSEQ * 6;
  float* gp = grams + (size_t)bh * TSEQ * NGRAM;
  float g = 0.f;
  int t0 = c * CHUNK;
  for (int k = 0; k < CHUNK; ++k) {
    int t = t0 + k;
    gp[(size_t)t * NGRAM + p] = g;
    float o = wlp[t * 6 + i] * wlp[t * 6 + j];
    g = DECAYF * g + o;
  }
  carry[u] = g;
}

__global__ void gram_scan_carry(float* __restrict__ carry)
{
  int s = blockIdx.x * blockDim.x + threadIdx.x;
  if (s >= BB * NH * NGRAM) return;
  float powL = __powf(DECAYF, (float)CHUNK);
  float M = 0.f;
  for (int c = 0; c < NCHUNK; ++c) {
    int idx = s * NCHUNK + c;
    float le = carry[idx];
    carry[idx] = M;
    M = powL * M + le;
  }
}

__global__ __launch_bounds__(128)
void gram_mlp(const float* __restrict__ grams, const float* __restrict__ carry,
              const float* __restrict__ m1w, const float* __restrict__ m1b,
              const float* __restrict__ m2w, const float* __restrict__ m2b,
              float* __restrict__ memout)
{
  __shared__ float s1[NGRAM * DH], sb1[DH], s2[DH * DH], sb2[DH];
  for (int i = threadIdx.x; i < NGRAM * DH; i += blockDim.x) s1[i] = m1w[i];
  for (int i = threadIdx.x; i < DH; i += blockDim.x) { sb1[i] = m1b[i]; sb2[i] = m2b[i]; }
  for (int i = threadIdx.x; i < DH * DH; i += blockDim.x) s2[i] = m2w[i];
  __syncthreads();

  int idx = blockIdx.x * blockDim.x + threadIdx.x;
  if (idx >= BB * NH * TSEQ) return;
  int t = idx % TSEQ, bh = idx / TSEQ;
  int chunk = t / CHUNK;
  float dec = __powf(DECAYF, (float)(t % CHUNK));

  float gf[NGRAM];
  #pragma unroll
  for (int p = 0; p < NGRAM; ++p)
    gf[p] = grams[(size_t)idx * NGRAM + p] +
            dec * carry[(bh * NGRAM + p) * NCHUNK + chunk];

  float hid[DH];
  #pragma unroll 4
  for (int j = 0; j < DH; ++j) {
    float v = sb1[j];
    #pragma unroll
    for (int p = 0; p < NGRAM; ++p) v += gf[p] * s1[p * DH + j];
    hid[j] = 0.5f * v * (1.f + erff(v * 0.70710678f));
  }
  #pragma unroll 4
  for (int d = 0; d < DH; ++d) {
    float v = sb2[d];
    #pragma unroll 8
    for (int j = 0; j < DH; ++j) v += hid[j] * s2[j * DH + d];
    memout[(size_t)idx * DH + d] = v;
  }
}

__global__ void combine_kernel(const float* __restrict__ seq,
                               const float* __restrict__ mem,
                               const float* __restrict__ gate,
                               float* __restrict__ comb)
{
  size_t idx = (size_t)blockIdx.x * blockDim.x + threadIdx.x;
  if (idx >= (size_t)BB * TSEQ * DM) return;
  int col = (int)(idx % DM);
  size_t bt = idx / DM;
  int h = col / DH, d = col % DH;
  int b = (int)(bt / TSEQ), t = (int)(bt % TSEQ);
  size_t sidx = (((size_t)b * NH + h) * TSEQ + t) * DH + d;
  comb[idx] = seq[sidx] + gate[bt * NH + h] * mem[sidx];
}

// ===========================================================================
// Launch orchestration
// ===========================================================================
extern "C" void kernel_launch(void* const* d_in, const int* in_sizes, int n_in,
                              void* d_out, int out_size, void* d_ws, size_t ws_size,
                              hipStream_t stream)
{
  const float* x      = (const float*)d_in[0];
  const float* qkv_w  = (const float*)d_in[1];
  const float* qkv_b  = (const float*)d_in[2];
  const float* w1_w   = (const float*)d_in[3];
  const float* w2_w   = (const float*)d_in[4];
  const float* mlp1_w = (const float*)d_in[5];
  const float* mlp1_b = (const float*)d_in[6];
  const float* mlp2_w = (const float*)d_in[7];
  const float* mlp2_b = (const float*)d_in[8];
  const float* gate_w = (const float*)d_in[9];
  const float* gate_b = (const float*)d_in[10];
  const float* out_w  = (const float*)d_in[11];
  const float* out_b  = (const float*)d_in[12];
  float* out = (float*)d_out;

  float* ws = (float*)d_ws;
  const size_t MROWS = (size_t)BB * TSEQ;            // 4096
  size_t off = 0;
  bf16_t* qkvh  = (bf16_t*)(ws + off); off += MROWS * 3 * DM / 2;   // bf16 halves
  float* wcat   = ws + off; off += (size_t)DM * 128;
  float* w12x   = ws + off; off += MROWS * 128;
  float* wl     = ws + off; off += (size_t)BB * NH * TSEQ * 6;
  float* grams  = ws + off; off += (size_t)BB * NH * TSEQ * NGRAM;
  float* carry  = ws + off; off += 16384;
  float* memout = ws + off; off += (size_t)BB * NH * TSEQ * DH;
  float* seqout = ws + off; off += (size_t)BB * NH * TSEQ * DH;
  float* gate   = ws + off; off += MROWS * NH;
  float* comb   = ws + off; off += MROWS * DM;

  // 1. QKV projection -> bf16 output consumed by attention
  gemm_bf16_wmma<<<dim3(3 * DM / 128, MROWS / 128), 256, 0, stream>>>(
      x, qkv_w, qkv_b, (void*)qkvh, (int)MROWS, 3 * DM, DM, 1, 1);

  // 2. fused w1|w2 projection (N=128, f32 out)
  concat_w12<<<(DM * DH + 255) / 256, 256, 0, stream>>>(w1_w, w2_w, wcat);
  gemm_bf16_wmma<<<dim3(1, MROWS / 128), 256, 0, stream>>>(
      x, wcat, nullptr, (void*)w12x, (int)MROWS, 128, DM, 0, 0);

  // 3. gate
  gate_kernel<<<(BB * TSEQ * NH + 255) / 256, 256, 0, stream>>>(x, gate_w, gate_b, gate);

  // 4. plucker
  plucker_kernel<<<(BB * NH * TSEQ + 255) / 256, 256, 0, stream>>>(w12x, wl);

  // 5-7. chunked decay scan + gram MLP
  gram_scan_local<<<(BB * NH * NGRAM * NCHUNK + 255) / 256, 256, 0, stream>>>(wl, grams, carry);
  gram_scan_carry<<<(BB * NH * NGRAM + 255) / 256, 256, 0, stream>>>(carry);
  gram_mlp<<<(BB * NH * TSEQ) / 128, 128, 0, stream>>>(
      grams, carry, mlp1_w, mlp1_b, mlp2_w, mlp2_b, memout);

  // 8. flash attention (bf16 QKV)
  flash_attn_wmma<<<dim3(TSEQ / 64, NH, BB), 128, 0, stream>>>(qkvh, seqout);

  // 9. combine
  combine_kernel<<<(int)((MROWS * DM) / 256), 256, 0, stream>>>(seqout, memout, gate, comb);

  // 10. output projection -> d_out (f32)
  gemm_bf16_wmma<<<dim3(DM / 128, MROWS / 128), 256, 0, stream>>>(
      comb, out_w, out_b, (void*)out, (int)MROWS, DM, DM, 1, 0);
}
